// EdgeModel_44418551775947
// MI455X (gfx1250) — matrix operand order repaired and checked by reading, not compiled, not run on hardware
//
#include <hip/hip_runtime.h>
#include <hip/hip_bf16.h>

// ---------------------------------------------------------------------------
// EdgeModel MLP on MI455X (gfx1250, wave32, WMMA):
//   x  = concat(src, dst, edge_attr, u[batch])      [E,512] fp32
//   h1 = softplus(x @ W1 + b1)                      [E,256]
//   h2 = softplus(h1 @ W2 + b2)                     [E,256]
//   y  = h2 @ W3 + b3                               [E,128] fp32
//
// bf16 WMMA (v_wmma_f32_16x16x32_bf16) with f32 accumulation.
// 64-edge tile per workgroup (halves L2 weight re-fetch vs 32-edge tile);
// each wave computes 2 m-subtiles x 4 n-subtiles so every B fragment is
// reused by two WMMAs and every A fragment by four.
// ---------------------------------------------------------------------------

typedef __attribute__((ext_vector_type(16))) __bf16  v16bf;
typedef __attribute__((ext_vector_type(8)))  float   v8f;
typedef __attribute__((ext_vector_type(4)))  unsigned int vu4;

union Frag {
    v16bf v;
    vu4   q[2];
};

static __device__ __forceinline__ unsigned short f2bf(float f) {
    // round-to-nearest-even fp32 -> bf16
    unsigned u = __builtin_bit_cast(unsigned, f);
    unsigned r = u + 0x7FFFu + ((u >> 16) & 1u);
    return (unsigned short)(r >> 16);
}

static __device__ __forceinline__ float softplus_ref(float x) {
    // torch Softplus(beta=1, threshold=20)
    return (x > 20.0f) ? x : __logf(1.0f + __expf(x));
}

// ---------------------------------------------------------------------------
// Weight conversion: f32 row-major W[K][N]  ->  bf16 transposed Wt[N][K]
//   ws[0          .. 131071] : Wt1  [256][512]
//   ws[131072     .. 196607] : Wt2  [256][256]
//   ws[196608     .. 229375] : Wt3  [128][256]
// ---------------------------------------------------------------------------
__global__ void edgemlp_convert_weights(const float* __restrict__ W1,
                                        const float* __restrict__ W2,
                                        const float* __restrict__ W3,
                                        unsigned short* __restrict__ ws) {
    int idx = blockIdx.x * blockDim.x + threadIdx.x;
    if (idx < 256 * 512) {                    // Wt1[n][k] = W1[k][n], K=512,N=256
        int n = idx >> 9, k = idx & 511;
        ws[idx] = f2bf(W1[(size_t)k * 256 + n]);
    } else if (idx < 256 * 512 + 256 * 256) { // Wt2[n][k] = W2[k][n], K=256,N=256
        int j = idx - 256 * 512;
        int n = j >> 8, k = j & 255;
        ws[idx] = f2bf(W2[(size_t)k * 256 + n]);
    } else if (idx < 229376) {                // Wt3[n][k] = W3[k][n], K=256,N=128
        int j = idx - (256 * 512 + 256 * 256);
        int n = j >> 8, k = j & 255;
        ws[idx] = f2bf(W3[(size_t)k * 128 + n]);
    }
}

// ---------------------------------------------------------------------------
// Per-wave register-blocked matmul:
//   acc[mi][ni] += A_mi(16x32 from LDS) x B_ni(32x16 from global Wt)
// A layout (ISA 16-bit A 16x32): lane<16 -> row m, K {8t..8t+7, 16+8t..23+8t}
// B layout: lane = N column, same K order as A: lane<16 K {0..7,16..23},
//   lanes 16-31 K {8..15,24..31}, read from Wt[N][K].
// ---------------------------------------------------------------------------
template <int MTILES, int NTILES, int KBLOCKS>
static __device__ __forceinline__ void mm_block(const unsigned short* __restrict__ As,
                                                int a_stride,
                                                const unsigned short* __restrict__ Wt,
                                                int wt_stride,
                                                int m_base, int n_base,
                                                int lane, v8f (*acc)[NTILES]) {
    const int l15 = lane & 15;
    const int t   = lane >> 4;
#pragma unroll
    for (int kb = 0; kb < KBLOCKS; ++kb) {
        const int k0 = kb * 32;
        Frag a[MTILES];
#pragma unroll
        for (int mi = 0; mi < MTILES; ++mi) {
            const unsigned short* arow = As + (m_base + mi * 16 + l15) * a_stride;
            a[mi].q[0] = *reinterpret_cast<const vu4*>(arow + k0 + 8 * t);
            a[mi].q[1] = *reinterpret_cast<const vu4*>(arow + k0 + 16 + 8 * t);
        }
#pragma unroll
        for (int i = 0; i < NTILES; ++i) {
            Frag b;
            const unsigned short* bp =
                Wt + (size_t)(n_base + i * 16 + l15) * wt_stride + k0 + 8 * t;
            b.q[0] = *reinterpret_cast<const vu4*>(bp);
            b.q[1] = *reinterpret_cast<const vu4*>(bp + 16);
#pragma unroll
            for (int mi = 0; mi < MTILES; ++mi) {
                acc[mi][i] = __builtin_amdgcn_wmma_f32_16x16x32_bf16(
                    false, a[mi].v, false, b.v, (short)0, acc[mi][i], false, false);
            }
        }
    }
}

// ---------------------------------------------------------------------------
// Main fused MLP kernel: one 256-thread block (8 waves) per 64-edge tile.
// LDS: X [64][520] bf16 (66560 B), H1 [64][264] (33792 B), H2 aliases X.
// +8-element row pads -> 4-bank shift per row -> conflict-free b128 reads.
// ---------------------------------------------------------------------------
#define XSTR 520
#define HSTR 264

__global__ void __launch_bounds__(256)
edgemlp_fused(const float* __restrict__ src,
              const float* __restrict__ dst,
              const float* __restrict__ eat,
              const float* __restrict__ u,
              const int*   __restrict__ batch,
              const float* __restrict__ b1,
              const float* __restrict__ b2,
              const float* __restrict__ b3,
              const unsigned short* __restrict__ Wt1,
              const unsigned short* __restrict__ Wt2,
              const unsigned short* __restrict__ Wt3,
              float* __restrict__ out,
              int E, int G) {
    __shared__ __align__(16) unsigned short smem[64 * XSTR + 64 * HSTR]; // 100352 B
    unsigned short* Xsh  = smem;                 // [64][XSTR]
    unsigned short* H1sh = smem + 64 * XSTR;     // [64][HSTR]
    unsigned short* H2sh = smem;                 // aliases X (dead after layer 1)

    const int m0      = blockIdx.x * 64;
    const int wave    = threadIdx.x >> 5;
    const int lane    = threadIdx.x & 31;
    const int l15     = lane & 15;
    const int t       = lane >> 4;
    const int mt_base = (wave >> 2) * 2;         // 0 or 2 : first of 2 m-subtiles

    // ---- Stage X = concat(src, dst, edge_attr, u[batch]) as bf16 into LDS ----
    for (int v = threadIdx.x; v < 64 * 128; v += 256) {
        const int r   = v >> 7;          // row in tile (0..63)
        const int c4  = v & 127;         // float4 group
        const int col = c4 * 4;
        const int er  = min(m0 + r, E - 1);
        const float* sp;
        if (col < 128) {
            sp = src + (size_t)er * 128 + col;
        } else if (col < 256) {
            sp = dst + (size_t)er * 128 + (col - 128);
        } else if (col < 384) {
            sp = eat + (size_t)er * 128 + (col - 256);
        } else {
            int g = batch[er];
            g = min(max(g, 0), G - 1);
            sp = u + (size_t)g * 128 + (col - 384);
        }
        const float4 f = *reinterpret_cast<const float4*>(sp);
        const unsigned lo = ((unsigned)f2bf(f.y) << 16) | f2bf(f.x);
        const unsigned hi = ((unsigned)f2bf(f.w) << 16) | f2bf(f.z);
        *reinterpret_cast<unsigned long long*>(&Xsh[r * XSTR + col]) =
            ((unsigned long long)hi << 32) | lo;
    }
    __syncthreads();

    // ---- Layer 1: H1 = softplus(X @ W1 + b1)   (K=512 -> 16 K-blocks) ----
    {
        const int n_base = (wave & 3) * 64;      // 4 n-tiles of 16
        v8f acc[2][4] = {};
        mm_block<2, 4, 16>(Xsh, XSTR, Wt1, 512, mt_base * 16, n_base, lane, acc);
#pragma unroll
        for (int mi = 0; mi < 2; ++mi) {
#pragma unroll
            for (int i = 0; i < 4; ++i) {
                const int n = n_base + i * 16 + l15;
                const float bias = b1[n];
#pragma unroll
                for (int e = 0; e < 8; ++e) {
                    const int m = (mt_base + mi) * 16 + t * 8 + e;
                    H1sh[m * HSTR + n] = f2bf(softplus_ref(acc[mi][i][e] + bias));
                }
            }
        }
    }
    __syncthreads();

    // ---- Layer 2: H2 = softplus(H1 @ W2 + b2)  (K=256 -> 8 K-blocks) ----
    {
        const int n_base = (wave & 3) * 64;
        v8f acc[2][4] = {};
        mm_block<2, 4, 8>(H1sh, HSTR, Wt2, 256, mt_base * 16, n_base, lane, acc);
#pragma unroll
        for (int mi = 0; mi < 2; ++mi) {
#pragma unroll
            for (int i = 0; i < 4; ++i) {
                const int n = n_base + i * 16 + l15;
                const float bias = b2[n];
#pragma unroll
                for (int e = 0; e < 8; ++e) {
                    const int m = (mt_base + mi) * 16 + t * 8 + e;
                    H2sh[m * HSTR + n] = f2bf(softplus_ref(acc[mi][i][e] + bias));
                }
            }
        }
    }
    __syncthreads();

    // ---- Layer 3: out = H2 @ W3 + b3  (K=256 -> 8 K-blocks, N=128) ----
    {
        const int n_base = (wave & 3) * 32;      // 2 n-tiles of 16
        v8f acc[2][2] = {};
        mm_block<2, 2, 8>(H2sh, HSTR, Wt3, 256, mt_base * 16, n_base, lane, acc);
#pragma unroll
        for (int mi = 0; mi < 2; ++mi) {
#pragma unroll
            for (int i = 0; i < 2; ++i) {
                const int n = n_base + i * 16 + l15;
                const float bias = b3[n];
#pragma unroll
                for (int e = 0; e < 8; ++e) {
                    const int m = (mt_base + mi) * 16 + t * 8 + e;
                    const int row = m0 + m;
                    if (row < E) out[(size_t)row * 128 + n] = acc[mi][i][e] + bias;
                }
            }
        }
    }
}

// ---------------------------------------------------------------------------
// Host launcher. Inputs (setup_inputs order):
//  0:src 1:dst 2:edge_attr 3:u 4:batch 5:W1 6:b1 7:W2 8:b2 9:W3 10:b3
// d_ws holds the bf16 transposed weights (458752 bytes needed).
// ---------------------------------------------------------------------------
extern "C" void kernel_launch(void* const* d_in, const int* in_sizes, int n_in,
                              void* d_out, int out_size, void* d_ws, size_t ws_size,
                              hipStream_t stream) {
    const float* src  = (const float*)d_in[0];
    const float* dst  = (const float*)d_in[1];
    const float* eat  = (const float*)d_in[2];
    const float* u    = (const float*)d_in[3];
    const int*   bat  = (const int*)d_in[4];
    const float* W1   = (const float*)d_in[5];
    const float* b1   = (const float*)d_in[6];
    const float* W2   = (const float*)d_in[7];
    const float* b2   = (const float*)d_in[8];
    const float* W3   = (const float*)d_in[9];
    const float* b3   = (const float*)d_in[10];
    float* out = (float*)d_out;

    const int E = in_sizes[0] / 128;   // edges
    const int G = in_sizes[3] / 128;   // graphs

    unsigned short* wt = (unsigned short*)d_ws;
    edgemlp_convert_weights<<<(229376 + 255) / 256, 256, 0, stream>>>(W1, W2, W3, wt);

    const int blocks = (E + 63) / 64;
    edgemlp_fused<<<blocks, 256, 0, stream>>>(
        src, dst, eat, u, bat, b1, b2, b3,
        wt,                  // Wt1 [256][512]
        wt + 131072,         // Wt2 [256][256]
        wt + 196608,         // Wt3 [128][256]
        out, E, G);
}
